// Model_57655640981586
// MI455X (gfx1250) — matrix operand order repaired and checked by reading, not compiled, not run on hardware
//
#include <hip/hip_runtime.h>
#include <stdint.h>

// ---------------------------------------------------------------------------
// Types / helpers
// ---------------------------------------------------------------------------
typedef __attribute__((ext_vector_type(16))) __bf16 v16bf;
typedef __attribute__((ext_vector_type(8)))  float  v8f;

struct Frag32B { uint4 lo; uint4 hi; };

__device__ __forceinline__ unsigned short f2bf(float f) {
  return __builtin_bit_cast(unsigned short, (__bf16)f);  // hw v_cvt, RNE
}
__device__ __forceinline__ __bf16 us2bf(unsigned short s) {
  return __builtin_bit_cast(__bf16, s);
}
// two aligned 16B loads -> one 16-element bf16 fragment
__device__ __forceinline__ v16bf load_frag(const unsigned short* p_lo,
                                           const unsigned short* p_hi) {
  Frag32B f;
  f.lo = *reinterpret_cast<const uint4*>(p_lo);
  f.hi = *reinterpret_cast<const uint4*>(p_hi);
  return __builtin_bit_cast(v16bf, f);
}

// ---------------------------------------------------------------------------
// CDNA5 async global->LDS copy (ASYNCcnt-tracked) with portable fallbacks
// ---------------------------------------------------------------------------
__device__ __forceinline__ void async_cp_f32(float* lds_dst, const float* gsrc) {
#if __has_builtin(__builtin_amdgcn_global_load_async_to_lds_b32)
  __builtin_amdgcn_global_load_async_to_lds_b32(
      (__attribute__((address_space(1))) int*)(uintptr_t)gsrc,
      (__attribute__((address_space(3))) int*)(unsigned)(uintptr_t)lds_dst,
      0, 0);
#elif defined(__AMDGCN__)
  unsigned ldso = (unsigned)(uintptr_t)lds_dst;
  asm volatile("global_load_async_to_lds_b32 %0, %1, off"
               :: "v"(ldso), "v"(gsrc) : "memory");
#else
  (void)lds_dst; (void)gsrc;
#endif
}

template <int N>
__device__ __forceinline__ void s_wait_async() {
#if __has_builtin(__builtin_amdgcn_s_wait_asynccnt)
  __builtin_amdgcn_s_wait_asynccnt(N);
#elif defined(__AMDGCN__)
  asm volatile("s_wait_asynccnt %0" :: "i"(N));
#endif
}

// ---------------------------------------------------------------------------
// small prep kernels
// ---------------------------------------------------------------------------
__global__ void cvt_bf16_kernel(const float* __restrict__ in,
                                unsigned short* __restrict__ out, int n) {
  int i = blockIdx.x * blockDim.x + threadIdx.x;
  if (i < n) out[i] = f2bf(in[i]);
}

__global__ void emb_norm_kernel(const float* __restrict__ emb,
                                float* __restrict__ e2) {
  int k = blockIdx.x * blockDim.x + threadIdx.x;
  if (k < 512) {
    float s = 0.f;
    for (int c = 0; c < 64; ++c) { float v = emb[k * 64 + c]; s += v * v; }
    e2[k] = s;
  }
}

// ---------------------------------------------------------------------------
// Encoder conv1: x(16,768,4096) * w(32,768,4) stride2 pad1 + bias, ReLU
// implicit GEMM: M=32 (co), K=3072 (ci*4+tap), N=2048 positions per batch.
// WG = 128 threads (4 waves); one batch, 64 positions per WG.
// K-loop: 96 chunks of 32 (8 input channels x 4 taps).
// Double-buffered LDS staging via async global->LDS copies:
//   - exactly 9 async instructions per wave per chunk, full EXEC always
//     (boundary-OOB lanes are redirected to a trash slot; the real pad slots
//     p=0 / p=129..131 are zeroed once in the prologue and, for boundary WGs,
//     never overwritten) so s_wait_asynccnt 9 provably drains chunk k after
//     chunk k+1 has been issued.
// ---------------------------------------------------------------------------
// per-chunk stage: rows = 8 input channels, cols = 132 positions (pitch 136
// with the two 16-lane halves of a fragment read 2 rows apart:
// 2*136 % 64 == 16).
__device__ __forceinline__ void stage_chunk(const float* __restrict__ xrows,
                                            float (*__restrict__ buf)[136],
                                            float* __restrict__ trash,
                                            int t0, int tid) {
  const int gbase = 2 * t0 - 1;
  // issues 0..7: row = it, position p = tid (0..127)
  #pragma unroll
  for (int it = 0; it < 8; ++it) {
    int g = gbase + tid;
    bool ok = (g >= 0) && (g < 4096);
    int gc = ok ? g : (g < 0 ? 0 : 4095);
    float* dst = ok ? &buf[it][tid] : &trash[tid];
    async_cp_f32(dst, xrows + (size_t)it * 4096 + gc);
  }
  // issue 8: tail cols p = 128 + wave; rows covered by lanes 0..7 of each wave
  {
    int wv = tid >> 5, ln = tid & 31;
    bool act = ln < 8;
    int row = ln & 7;
    int p = 128 + wv;
    int g = gbase + p;               // >= 127, so only upper bound matters
    bool ok = act && (g < 4096);
    int gc = ok ? g : 4095;
    int rowc = ok ? row : 0;
    float* dst = ok ? &buf[rowc][p] : &trash[tid];
    async_cp_f32(dst, xrows + (size_t)rowc * 4096 + gc);
  }
}

__global__ __launch_bounds__(128)
void conv1_wmma_kernel(const float* __restrict__ x,
                       const unsigned short* __restrict__ wbf,  // (32,768,4) bf16
                       const float* __restrict__ bias,
                       float* __restrict__ out)                 // (16,32,2048)
{
  __shared__ float sbuf[2][8][136];
  __shared__ float strash[128];

  const int n    = blockIdx.y;
  const int t0   = blockIdx.x * 64;
  const int tid  = threadIdx.x;
  const int lane = tid & 31;
  const int wave = tid >> 5;
  const int hi   = lane >> 4;
  const int l15  = lane & 15;
  const int tl   = wave * 16 + l15;   // local output column (0..63)

  // prologue: zero the pad slots once (p=0, p=129..131; both buffers).
  // Interior WGs overwrite them with real data every chunk; boundary WGs'
  // OOB lanes are permanently redirected to strash, so the zeros persist.
  if (tid < 64) {
    int b   = tid >> 5;
    int rem = tid & 31;
    int row = rem >> 2;
    int cs  = rem & 3;
    int p   = (cs == 0) ? 0 : 128 + cs;
    sbuf[b][row][p] = 0.f;
  }
  __syncthreads();

  const float* xb = x + (size_t)n * 768 * 4096;
  stage_chunk(xb, sbuf[0], strash, t0, tid);   // chunk 0 -> buffer 0

  v8f acc0 = {};
  v8f acc1 = {};

  for (int kc = 0; kc < 96; ++kc) {
    // prefetch chunk kc+1 into the other buffer, then drain chunk kc
    if (kc + 1 < 96) {
      stage_chunk(xb + (size_t)(kc + 1) * 8 * 4096, sbuf[(kc + 1) & 1],
                  strash, t0, tid);
      s_wait_async<9>();
    } else {
      s_wait_async<0>();
    }
    __syncthreads();   // all waves' chunk-kc copies have landed

    // A fragments (weights bf16): two M tiles (co 0..15 and 16..31)
    // element e -> K = kc*32 + (e&7) + 16*(e>>3) + 8*hi  (two contig 8-runs)
    const int kbase = kc * 32;
    const unsigned short* wrow0 = wbf + (size_t)l15        * 3072 + kbase + 8 * hi;
    const unsigned short* wrow1 = wbf + (size_t)(l15 + 16) * 3072 + kbase + 8 * hi;
    v16bf a0 = load_frag(wrow0, wrow0 + 16);
    v16bf a1 = load_frag(wrow1, wrow1 + 16);

    // B fragment from LDS (f32 -> bf16 at read): elem e -> K = (e&7) + 16*(e>>3) + 8*hi
    const float* bufp = &sbuf[kc & 1][0][0];
    v16bf bfr;
    #pragma unroll
    for (int e = 0; e < 16; ++e) {
      int kk  = (e & 7) + 16 * (e >> 3) + 8 * hi;
      int cil = kk >> 2;
      int kt  = kk & 3;
      bfr[e] = (__bf16)bufp[cil * 136 + 2 * tl + kt];
    }

    acc0 = __builtin_amdgcn_wmma_f32_16x16x32_bf16(false, a0, false, bfr,
                                                   (short)0, acc0, false, false);
    acc1 = __builtin_amdgcn_wmma_f32_16x16x32_bf16(false, a1, false, bfr,
                                                   (short)0, acc1, false, false);
    __syncthreads();   // reads done: safe for iter kc+1 to overwrite buf[kc&1]
  }

  // epilogue: bias + ReLU + store. C layout: elem r -> co = r + 8*hi, col = l15
  const int t = t0 + wave * 16 + l15;
  #pragma unroll
  for (int r = 0; r < 8; ++r) {
    int co = r + 8 * hi;
    float v0 = fmaxf(acc0[r] + bias[co],      0.f);
    float v1 = fmaxf(acc1[r] + bias[co + 16], 0.f);
    out[((size_t)(n * 32 + co))      * 2048 + t] = v0;
    out[((size_t)(n * 32 + co + 16)) * 2048 + t] = v1;
  }
}

// ---------------------------------------------------------------------------
// generic f32 direct conv1d: out(b,co,t) (+residual)(+relu in/out)
// ---------------------------------------------------------------------------
__global__ void conv1d_generic(const float* __restrict__ in,
                               const float* __restrict__ w,
                               const float* __restrict__ bias,
                               const float* __restrict__ res,
                               float* __restrict__ out,
                               int B, int Cin, int Cout, int Lin, int Lout,
                               int K, int stride, int pad,
                               int relu_in, int relu_out)
{
  int idx = blockIdx.x * blockDim.x + threadIdx.x;
  int total = B * Cout * Lout;
  if (idx >= total) return;
  int t  = idx % Lout;
  int co = (idx / Lout) % Cout;
  int b  = idx / (Lout * Cout);
  float acc = bias ? bias[co] : 0.f;
  const float* ib = in + (size_t)b * Cin * Lin;
  const float* wb = w  + (size_t)co * Cin * K;
  for (int ci = 0; ci < Cin; ++ci) {
    const float* ic = ib + (size_t)ci * Lin;
    for (int k = 0; k < K; ++k) {
      int p = t * stride - pad + k;
      if (p >= 0 && p < Lin) {
        float v = ic[p];
        if (relu_in) v = fmaxf(v, 0.f);
        acc += v * wb[ci * K + k];
      }
    }
  }
  if (res) acc += res[idx];
  if (relu_out) acc = fmaxf(acc, 0.f);
  out[idx] = acc;
}

// ---------------------------------------------------------------------------
// generic f32 transposed conv1d (torch weight layout (Cin,Cout,K))
// out[b,o,l] = bias[o] + sum_{ci,k : l = t*stride - pad + k} in[b,ci,t]*w[ci,o,k]
// ---------------------------------------------------------------------------
__global__ void tconv1d_generic(const float* __restrict__ in,
                                const float* __restrict__ w,
                                const float* __restrict__ bias,
                                float* __restrict__ out,
                                int B, int Cin, int Cout, int Lin, int Lout,
                                int K, int stride, int pad,
                                int relu_in, int relu_out)
{
  int idx = blockIdx.x * blockDim.x + threadIdx.x;
  int total = B * Cout * Lout;
  if (idx >= total) return;
  int l = idx % Lout;
  int o = (idx / Lout) % Cout;
  int b = idx / (Lout * Cout);
  float acc = bias[o];
  for (int ci = 0; ci < Cin; ++ci) {
    const float* ic = in + ((size_t)b * Cin + ci) * Lin;
    const float* wc = w  + ((size_t)ci * Cout + o) * K;
    for (int k = 0; k < K; ++k) {
      int num = l + pad - k;
      if (num >= 0 && (num % stride) == 0) {
        int t = num / stride;
        if (t < Lin) {
          float v = ic[t];
          if (relu_in) v = fmaxf(v, 0.f);
          acc += v * wc[k];
        }
      }
    }
  }
  if (relu_out) acc = fmaxf(acc, 0.f);
  out[idx] = acc;
}

// ---------------------------------------------------------------------------
// pre-VQ 1x1 conv (fused ReLU on input) writing z transposed to (B*T,64) bf16
// ---------------------------------------------------------------------------
__global__ void prevq_kernel(const float* __restrict__ h,      // (16,64,1024)
                             const float* __restrict__ w,      // (64,64,1)
                             const float* __restrict__ bias,
                             unsigned short* __restrict__ zT)  // (16384,64) bf16
{
  int idx = blockIdx.x * blockDim.x + threadIdx.x;  // b*1024 + t
  if (idx >= 16384) return;
  int b = idx >> 10;
  int t = idx & 1023;
  float hin[64];
  #pragma unroll
  for (int ci = 0; ci < 64; ++ci)
    hin[ci] = fmaxf(h[((size_t)(b * 64 + ci)) * 1024 + t], 0.f);
  unsigned short* zr = zT + (size_t)idx * 64;
  for (int co = 0; co < 64; ++co) {
    float s = bias[co];
    #pragma unroll
    for (int ci = 0; ci < 64; ++ci) s += hin[ci] * w[co * 64 + ci];
    zr[co] = f2bf(s);
  }
}

// ---------------------------------------------------------------------------
// VQ: scores = z @ emb^T via WMMA (M=16 rows/wave, N=512 codes, K=64),
// argmin over (||e||^2 - 2*score), gather codebook row -> q (16,64,1024) f32.
// ---------------------------------------------------------------------------
__global__ __launch_bounds__(128)
void vq_wmma_kernel(const unsigned short* __restrict__ zT,     // (16384,64) bf16
                    const unsigned short* __restrict__ embbf,  // (512,64) bf16
                    const float* __restrict__ e2,               // (512)
                    const float* __restrict__ embf,             // (512,64) f32
                    float* __restrict__ q)                      // (16,64,1024)
{
  const int tid  = threadIdx.x;
  const int lane = tid & 31;
  const int wave = tid >> 5;
  const int hi   = lane >> 4;
  const int l15  = lane & 15;
  const int row0 = blockIdx.x * 64 + wave * 16;

  // A fragments: rows row0..row0+15, K(channel)=64 in two chunks of 32
  const unsigned short* ab = zT + (size_t)(row0 + l15) * 64;
  v16bf a0 = load_frag(ab + 8 * hi,      ab + 16 + 8 * hi);
  v16bf a1 = load_frag(ab + 32 + 8 * hi, ab + 48 + 8 * hi);

  float minv[8];
  int   mini[8];
  #pragma unroll
  for (int r = 0; r < 8; ++r) { minv[r] = 3.4e38f; mini[r] = 0; }

  for (int nt = 0; nt < 32; ++nt) {
    const int code = nt * 16 + l15;   // B column N = l15
    const unsigned short* bb = embbf + (size_t)code * 64;
    // B layout: element e -> K = kc*32 + (e&7) + 16*(e>>3) + 8*hi (two contig 8-runs)
    v16bf b0 = load_frag(bb + 8 * hi,      bb + 16 + 8 * hi);
    v16bf b1 = load_frag(bb + 32 + 8 * hi, bb + 48 + 8 * hi);
    v8f acc = {};
    acc = __builtin_amdgcn_wmma_f32_16x16x32_bf16(false, a0, false, b0,
                                                  (short)0, acc, false, false);
    acc = __builtin_amdgcn_wmma_f32_16x16x32_bf16(false, a1, false, b1,
                                                  (short)0, acc, false, false);
    const float e2n = e2[code];
    #pragma unroll
    for (int r = 0; r < 8; ++r) {
      float d = e2n - 2.f * acc[r];   // ||z||^2 constant per row: dropped
      if (d < minv[r]) { minv[r] = d; mini[r] = code; }
    }
  }

  // butterfly argmin across the 16 lanes sharing each row group
  #pragma unroll
  for (int off = 1; off < 16; off <<= 1) {
    #pragma unroll
    for (int r = 0; r < 8; ++r) {
      float ov = __shfl_xor(minv[r], off, 32);
      int   oi = __shfl_xor(mini[r], off, 32);
      if (ov < minv[r] || (ov == minv[r] && oi < mini[r])) {
        minv[r] = ov; mini[r] = oi;
      }
    }
  }

  // straight-through forward value == selected codebook row; gather to q
  #pragma unroll
  for (int r = 0; r < 8; ++r) {
    int row = row0 + r + 8 * hi;      // C elem r -> M = r + 8*hi
    int b = row >> 10;
    int t = row & 1023;
    const float* er = embf + (size_t)mini[r] * 64;
    for (int c = l15; c < 64; c += 16)
      q[((size_t)(b * 64 + c)) * 1024 + t] = er[c];
  }
}

// ---------------------------------------------------------------------------
// launch
// ---------------------------------------------------------------------------
extern "C" void kernel_launch(void* const* d_in, const int* in_sizes, int n_in,
                              void* d_out, int out_size, void* d_ws, size_t ws_size,
                              hipStream_t stream)
{
  (void)in_sizes; (void)n_in; (void)out_size; (void)ws_size;

  const float* x         = (const float*)d_in[0];
  const float* enc_w1    = (const float*)d_in[1];
  const float* enc_b1    = (const float*)d_in[2];
  const float* enc_w2    = (const float*)d_in[3];
  const float* enc_b2    = (const float*)d_in[4];
  const float* enc_w3    = (const float*)d_in[5];
  const float* enc_b3    = (const float*)d_in[6];
  const float* enc_r0_w1 = (const float*)d_in[7];
  const float* enc_r0_w2 = (const float*)d_in[8];
  const float* enc_r1_w1 = (const float*)d_in[9];
  const float* enc_r1_w2 = (const float*)d_in[10];
  const float* prevq_w   = (const float*)d_in[11];
  const float* prevq_b   = (const float*)d_in[12];
  const float* emb       = (const float*)d_in[13];
  const float* dec_w1    = (const float*)d_in[14];
  const float* dec_b1    = (const float*)d_in[15];
  const float* dec_r0_w1 = (const float*)d_in[16];
  const float* dec_r0_w2 = (const float*)d_in[17];
  const float* dec_r1_w1 = (const float*)d_in[18];
  const float* dec_r1_w2 = (const float*)d_in[19];
  const float* dect1_w   = (const float*)d_in[20];
  const float* dect1_b   = (const float*)d_in[21];
  const float* dect2_w   = (const float*)d_in[22];
  const float* dect2_b   = (const float*)d_in[23];

  char* ws = (char*)d_ws;
  size_t off = 0;
  auto alloc = [&](size_t bytes) -> void* {
    void* p = ws + off;
    off = (off + bytes + 255) & ~(size_t)255;
    return p;
  };

  unsigned short* wbf   = (unsigned short*)alloc((size_t)32 * 768 * 4 * 2);
  unsigned short* embbf = (unsigned short*)alloc((size_t)512 * 64 * 2);
  float*          e2    = (float*)alloc(512 * sizeof(float));
  float*          h1    = (float*)alloc((size_t)16 * 32 * 2048 * 4);  // conv1 out
  float*          hA    = (float*)alloc((size_t)16 * 64 * 1024 * 4);
  float*          hB    = (float*)alloc((size_t)16 * 64 * 1024 * 4);
  float*          t1    = (float*)alloc((size_t)16 * 32 * 1024 * 4);  // res branch
  unsigned short* zT    = (unsigned short*)alloc((size_t)16384 * 64 * 2);
  float*          q     = (float*)alloc((size_t)16 * 64 * 1024 * 4);
  float*          d1    = (float*)alloc((size_t)16 * 32 * 2048 * 4);

  // weight / codebook conversion + codebook norms
  cvt_bf16_kernel<<<(98304 + 255) / 256, 256, 0, stream>>>(enc_w1, wbf, 98304);
  cvt_bf16_kernel<<<(32768 + 255) / 256, 256, 0, stream>>>(emb, embbf, 32768);
  emb_norm_kernel<<<2, 256, 0, stream>>>(emb, e2);

  // encoder
  conv1_wmma_kernel<<<dim3(32, 16), 128, 0, stream>>>(x, wbf, enc_b1, h1);

  auto conv = [&](const float* in, const float* w, const float* bias,
                  const float* res, float* out,
                  int B, int Cin, int Cout, int Lin, int Lout,
                  int K, int s, int p, int ri, int ro) {
    int total = B * Cout * Lout;
    conv1d_generic<<<(total + 255) / 256, 256, 0, stream>>>(
        in, w, bias, res, out, B, Cin, Cout, Lin, Lout, K, s, p, ri, ro);
  };

  conv(h1, enc_w2, enc_b2, nullptr, hA, 16, 32, 64, 2048, 1024, 4, 2, 1, 0, 1);
  conv(hA, enc_w3, enc_b3, nullptr, hB, 16, 64, 64, 1024, 1024, 3, 1, 1, 0, 0);
  // encoder residual block 0
  conv(hB, enc_r0_w1, nullptr, nullptr, t1, 16, 64, 32, 1024, 1024, 3, 1, 1, 1, 0);
  conv(t1, enc_r0_w2, nullptr, hB,      hA, 16, 32, 64, 1024, 1024, 1, 1, 0, 1, 0);
  // encoder residual block 1
  conv(hA, enc_r1_w1, nullptr, nullptr, t1, 16, 64, 32, 1024, 1024, 3, 1, 1, 1, 0);
  conv(t1, enc_r1_w2, nullptr, hA,      hB, 16, 32, 64, 1024, 1024, 1, 1, 0, 1, 0);
  // stack's trailing ReLU is fused into prevq's input read

  prevq_kernel<<<64, 256, 0, stream>>>(hB, prevq_w, prevq_b, zT);
  vq_wmma_kernel<<<256, 128, 0, stream>>>(zT, embbf, e2, emb, q);

  // decoder
  conv(q,  dec_w1, dec_b1, nullptr, hA, 16, 64, 64, 1024, 1024, 3, 1, 1, 0, 0);
  conv(hA, dec_r0_w1, nullptr, nullptr, t1, 16, 64, 32, 1024, 1024, 3, 1, 1, 1, 0);
  conv(t1, dec_r0_w2, nullptr, hA,      hB, 16, 32, 64, 1024, 1024, 1, 1, 0, 1, 0);
  conv(hB, dec_r1_w1, nullptr, nullptr, t1, 16, 64, 32, 1024, 1024, 3, 1, 1, 1, 0);
  conv(t1, dec_r1_w2, nullptr, hB,      hA, 16, 32, 64, 1024, 1024, 1, 1, 0, 1, 0);
  // stack's trailing ReLU fused into dect1's input read

  tconv1d_generic<<<(16 * 32 * 2048 + 255) / 256, 256, 0, stream>>>(
      hA, dect1_w, dect1_b, d1, 16, 64, 32, 1024, 2048, 4, 2, 1, 1, 1);
  tconv1d_generic<<<(16 * 64 * 4096 + 255) / 256, 256, 0, stream>>>(
      d1, dect2_w, dect2_b, (float*)d_out, 16, 32, 64, 2048, 4096, 4, 2, 1, 0, 0);
}
